// DkNNModel_62027917689406
// MI455X (gfx1250) — matrix operand order repaired and checked by reading, not compiled, not run on hardware
//
#include <hip/hip_runtime.h>
#include <stdint.h>

typedef __attribute__((ext_vector_type(16))) _Float16 v16h;
typedef __attribute__((ext_vector_type(8)))  float    v8f;
typedef __attribute__((ext_vector_type(2)))  __fp16   h2v;   // matches cvt_pkrtz return type

#define NB_CLASSES 10
#define DIMK 256        // feature dim (K), fixed by problem
#define KC   8          // 256 / 32 K-chunks per WMMA sweep
#define SLABS 256       // N-slab parallelism (waves per M-group)

static __device__ __forceinline__ float sum8sq(float4 a, float4 b) {
  return a.x*a.x + a.y*a.y + a.z*a.z + a.w*a.w
       + b.x*b.x + b.y*b.y + b.z*b.z + b.w*b.w;
}

// ---------------- Phase 1a: xqh = f16( x/||x|| - center ) ----------------
__global__ void prep_queries_kernel(const float* __restrict__ x,
                                    const float* __restrict__ center,
                                    _Float16* __restrict__ xqh, int B) {
  const int wid  = (int)((blockIdx.x * blockDim.x + threadIdx.x) >> 5);
  const int lane = (int)(threadIdx.x & 31);
  if (wid >= B) return;
  const float* row = x + (size_t)wid * DIMK;
  const int e = lane * 8;
  float4 a0 = *(const float4*)(row + e);
  float4 a1 = *(const float4*)(row + e + 4);
  float ss = sum8sq(a0, a1);
  #pragma unroll
  for (int off = 16; off >= 1; off >>= 1) ss += __shfl_xor(ss, off, 32);
  const float inv = 1.0f / sqrtf(ss);
  float4 c0 = *(const float4*)(center + e);
  float4 c1 = *(const float4*)(center + e + 4);
  union { uint4 q; h2v h[4]; } u;
  u.h[0] = __builtin_amdgcn_cvt_pkrtz(a0.x*inv - c0.x, a0.y*inv - c0.y);
  u.h[1] = __builtin_amdgcn_cvt_pkrtz(a0.z*inv - c0.z, a0.w*inv - c0.w);
  u.h[2] = __builtin_amdgcn_cvt_pkrtz(a1.x*inv - c1.x, a1.y*inv - c1.y);
  u.h[3] = __builtin_amdgcn_cvt_pkrtz(a1.z*inv - c1.z, a1.w*inv - c1.w);
  *(uint4*)(xqh + (size_t)wid * DIMK + e) = u.q;
}

// ---------------- Phase 1b: halfnorm[n] = 0.5 * ||X_n||^2 ----------------
__global__ void prep_halfnorm_kernel(const float* __restrict__ X,
                                     float* __restrict__ hn, int N) {
  const int wid  = (int)((blockIdx.x * blockDim.x + threadIdx.x) >> 5);
  const int lane = (int)(threadIdx.x & 31);
  if (wid >= N) return;
  const float* row = X + (size_t)wid * DIMK;
  const int e = lane * 8;
  float4 a0 = *(const float4*)(row + e);
  float4 a1 = *(const float4*)(row + e + 4);
  float ss = sum8sq(a0, a1);
  #pragma unroll
  for (int off = 16; off >= 1; off >>= 1) ss += __shfl_xor(ss, off, 32);
  if (lane == 0) hn[wid] = 0.5f * ss;
}

// ---------------- Phase 1c: init argmax cells (ws is poisoned) ----------------
__global__ void init_best_kernel(unsigned long long* __restrict__ best, int B) {
  int i = (int)(blockIdx.x * blockDim.x + threadIdx.x);
  if (i < B) best[i] = 0ull;
}

// ---------------- Phase 2: WMMA GEMM + fused argmax over train set ----------------
// Wave owns M=32 rows (A fully resident in VGPRs for K=256), strides N-tiles of 16.
__global__ __launch_bounds__(256, 1)
void knn_score_argmax_kernel(const _Float16* __restrict__ xqh,
                             const float* __restrict__ X,
                             const float* __restrict__ hn,
                             unsigned long long* __restrict__ best,
                             int B, int N) {
  const int lane = (int)(threadIdx.x & 31);
  const int wid  = (int)((blockIdx.x * blockDim.x + threadIdx.x) >> 5);
  const int nMGroups = B >> 5;               // B/32 m-groups
  const int g = wid % nMGroups;
  const int s = wid / nMGroups;              // slab id in [0, SLABS)
  const int mBase = g << 5;
  const int half = lane >> 4;                // 0: K-low half, 1: K-high half
  const int l15  = lane & 15;

  // --- preload all A fragments: 2 m-frags x 8 k-chunks (xqh is L2-hot) ---
  v16h aF[KC][2];
  {
    const int sel = half << 3;               // lanes>=16 hold K+16 (contiguous, matches B)
    #pragma unroll
    for (int f = 0; f < 2; ++f) {
      const _Float16* arow = xqh + (size_t)(mBase + (f << 4) + l15) * DIMK;
      #pragma unroll
      for (int kc = 0; kc < KC; ++kc) {
        union { v16h v; uint4 q[2]; } u;
        u.q[0] = *(const uint4*)(arow + kc * 32 + 2 * sel);        // K = k0+2*sel..+7
        u.q[1] = *(const uint4*)(arow + kc * 32 + 2 * sel + 8);    // K = k0+2*sel+8..+7
        aF[kc][f] = u.v;
      }
    }
  }

  float    bestv[2][8];
  unsigned bestn[2][8];
  #pragma unroll
  for (int f = 0; f < 2; ++f)
    #pragma unroll
    for (int r = 0; r < 8; ++r) { bestv[f][r] = -__builtin_inff(); bestn[f][r] = 0u; }

  const int nTiles = N >> 4;
  for (int t = s; t < nTiles; t += SLABS) {
    const int n = (t << 4) + l15;            // this lane's train row (B-matrix column)
    const float* brow = X + (size_t)n * DIMK + (half << 4);
    if (t + SLABS < nTiles)                  // speculative prefetch of next tile's row
      __builtin_prefetch(brow + (size_t)(SLABS << 4) * DIMK, 0, 1);

    v8f acc0 = {}; v8f acc1 = {};
    #pragma unroll
    for (int kc = 0; kc < KC; ++kc) {
      const float* p = brow + kc * 32;
      float4 f0 = *(const float4*)(p + 0);
      float4 f1 = *(const float4*)(p + 4);
      float4 f2 = *(const float4*)(p + 8);
      float4 f3 = *(const float4*)(p + 12);
      union { v16h v; h2v h[8]; } bu;        // B frag: 16 consecutive K per lane
      bu.h[0] = __builtin_amdgcn_cvt_pkrtz(f0.x, f0.y);
      bu.h[1] = __builtin_amdgcn_cvt_pkrtz(f0.z, f0.w);
      bu.h[2] = __builtin_amdgcn_cvt_pkrtz(f1.x, f1.y);
      bu.h[3] = __builtin_amdgcn_cvt_pkrtz(f1.z, f1.w);
      bu.h[4] = __builtin_amdgcn_cvt_pkrtz(f2.x, f2.y);
      bu.h[5] = __builtin_amdgcn_cvt_pkrtz(f2.z, f2.w);
      bu.h[6] = __builtin_amdgcn_cvt_pkrtz(f3.x, f3.y);
      bu.h[7] = __builtin_amdgcn_cvt_pkrtz(f3.z, f3.w);
      acc0 = __builtin_amdgcn_wmma_f32_16x16x32_f16(false, aF[kc][0], false, bu.v,
                                                    (short)0, acc0, false, false);
      acc1 = __builtin_amdgcn_wmma_f32_16x16x32_f16(false, aF[kc][1], false, bu.v,
                                                    (short)0, acc1, false, false);
    }

    const float h = hn[n];
    #pragma unroll
    for (int r = 0; r < 8; ++r) {
      float s0 = acc0[r] - h;                // score = dot - 0.5||Xn||^2
      float s1 = acc1[r] - h;
      if (s0 > bestv[0][r]) { bestv[0][r] = s0; bestn[0][r] = (unsigned)n; }
      if (s1 > bestv[1][r]) { bestv[1][r] = s1; bestn[1][r] = (unsigned)n; }
    }
  }

  // --- reduce over the 16 lanes of each half, then global u64 atomicMax ---
  #pragma unroll
  for (int f = 0; f < 2; ++f) {
    #pragma unroll
    for (int r = 0; r < 8; ++r) {
      unsigned vb = __float_as_uint(bestv[f][r]);
      vb = (vb & 0x80000000u) ? ~vb : (vb | 0x80000000u);   // order-preserving map
      unsigned long long key =
          ((unsigned long long)vb << 32) | (unsigned)(~bestn[f][r]); // ~n: ties->min n
      #pragma unroll
      for (int off = 8; off >= 1; off >>= 1) {              // stays within 16-lane half
        unsigned lo  = (unsigned)key, hi = (unsigned)(key >> 32);
        unsigned lo2 = __shfl_xor(lo, off, 32);
        unsigned hi2 = __shfl_xor(hi, off, 32);
        unsigned long long k2 = ((unsigned long long)hi2 << 32) | lo2;
        if (k2 > key) key = k2;
      }
      if (l15 == 0) {
        const int m = mBase + (f << 4) + (half << 3) + r;   // C layout: M = r + 8*half
        atomicMax(best + m, key);
      }
    }
  }
}

// ---------------- Phase 3: labels -> bincount -> searchsorted -> creds ----------------
__global__ void conformal_kernel(const unsigned long long* __restrict__ best,
                                 const int* __restrict__ labels,
                                 const int* __restrict__ tni,
                                 const int* __restrict__ cali,
                                 float* __restrict__ out,
                                 int B, int KNB1, int ncali) {
  const int i = (int)(blockIdx.x * blockDim.x + threadIdx.x);
  if (i >= B) return;
  const unsigned long long key = best[i];
  const unsigned n = ~(unsigned)(key & 0xFFFFFFFFull);      // decode winner index

  int counts[NB_CLASSES];
  #pragma unroll
  for (int c = 0; c < NB_CLASSES; ++c) counts[c] = 0;
  counts[labels[n]]++;
  const int* nrow = tni + (size_t)n * KNB1;
  for (int j = 0; j < KNB1; ++j) counts[labels[nrow[j]]]++;

  const int K = KNB1 + 1;
  float bestp = -1.0f; int bc = 0;
  #pragma unroll
  for (int c = 0; c < NB_CLASSES; ++c) {
    const int v = K - counts[c];                            // knns NOT in class c
    int lo = 0, hi = ncali;                                 // bisect_left
    while (lo < hi) { int mid = (lo + hi) >> 1; if (cali[mid] < v) lo = mid + 1; else hi = mid; }
    const float pc = (float)(ncali - lo) / (float)ncali;
    if (pc > bestp) { bestp = pc; bc = c; }                 // first-max like argmax
  }
  #pragma unroll
  for (int c = 0; c < NB_CLASSES; ++c)
    out[(size_t)i * NB_CLASSES + c] = (c == bc) ? bestp : 0.0f;
}

extern "C" void kernel_launch(void* const* d_in, const int* in_sizes, int n_in,
                              void* d_out, int out_size, void* d_ws, size_t ws_size,
                              hipStream_t stream) {
  const float* x      = (const float*)d_in[0];
  const float* X      = (const float*)d_in[1];
  const float* center = (const float*)d_in[2];
  const int*   labels = (const int*)d_in[3];
  const int*   tni    = (const int*)d_in[4];
  const int*   cali   = (const int*)d_in[5];

  const int d     = in_sizes[2];          // 256
  const int B     = in_sizes[0] / d;      // 1024
  const int N     = in_sizes[3];          // 100000
  const int KNB1  = in_sizes[4] / N;      // 74
  const int ncali = in_sizes[5];          // 1000
  (void)n_in; (void)out_size; (void)ws_size; (void)d;

  // workspace layout (≈ 0.93 MB total)
  char* ws = (char*)d_ws;
  _Float16* xqh = (_Float16*)ws;
  size_t off = (((size_t)B * DIMK * sizeof(_Float16)) + 255) & ~(size_t)255;
  float* hn = (float*)(ws + off);
  off += (((size_t)N * sizeof(float)) + 255) & ~(size_t)255;
  unsigned long long* best = (unsigned long long*)(ws + off);

  prep_queries_kernel<<<(B * 32 + 255) / 256, 256, 0, stream>>>(x, center, xqh, B);
  prep_halfnorm_kernel<<<(N * 32 + 255) / 256, 256, 0, stream>>>(X, hn, N);
  init_best_kernel<<<(B + 255) / 256, 256, 0, stream>>>(best, B);

  const int nMGroups   = B >> 5;              // 32
  const int totalWaves = nMGroups * SLABS;    // 8192
  knn_score_argmax_kernel<<<totalWaves / 8, 256, 0, stream>>>(xqh, X, hn, best, B, N);

  conformal_kernel<<<(B + 255) / 256, 256, 0, stream>>>(best, labels, tni, cali,
                                                        (float*)d_out, B, KNB1, ncali);
}
